// EnhancedGCN_56521769616160
// MI455X (gfx1250) — matrix operand order repaired and checked by reading, not compiled, not run on hardware
//
#include <hip/hip_runtime.h>

// ---------------------------------------------------------------------------
// EnhancedGCN forward for MI455X (gfx1250, wave32, WMMA)
//   h = in_feat @ lin_w.T + lin_b
//   deg = (#in-edges) + 1 ;  dis = rsqrt(deg) ; deg_inv = 1/deg
//   loop P=2: [res+LN+relu] ; xw = h @ conv_w.T ;
//             h_msg = conv_b + deg_inv*xw (self loop) + sum_e dis[r]dis[c]*xw[c]
//             h = h_msg + relu(h + root) * deg_inv
// GEMMs use v_wmma_f32_16x16x32_bf16 (bf16 in, f32 acc); A tiles are staged in
// LDS pre-swizzled into the WMMA operand layout; weights are pre-converted to
// bf16 once per launch.  Sparse scatter uses L2-resident f32 atomics.
// Assumes N % 16 == 0 (reference: N = 50000 = 3125 * 16).
// ---------------------------------------------------------------------------

typedef __attribute__((ext_vector_type(16))) __bf16 v16bf;
typedef __attribute__((ext_vector_type(4)))  __bf16 bf4;
typedef __attribute__((ext_vector_type(8)))  float  v8f;

constexpr int   Hc     = 128;
constexpr int   INc    = 256;
constexpr float LN_EPS = 1e-5f;

__device__ __forceinline__ __bf16 to_bf(float x) { return (__bf16)x; }

// ---- degree pipeline -------------------------------------------------------
__global__ void k_init_deg(float* __restrict__ deg, int n) {
  int i = blockIdx.x * blockDim.x + threadIdx.x;
  if (i < n) deg[i] = 1.0f;                      // +1 self-loop trick
}

__global__ void k_count_deg(const int* __restrict__ row, float* __restrict__ deg, int e) {
  int i = blockIdx.x * blockDim.x + threadIdx.x;
  if (i < e) atomicAdd(&deg[row[i]], 1.0f);
}

__global__ void k_finish_deg(const float* __restrict__ deg,
                             float* __restrict__ dis, float* __restrict__ dinv, int n) {
  int i = blockIdx.x * blockDim.x + threadIdx.x;
  if (i < n) {
    float d = deg[i];                            // d >= 1 always
    dinv[i] = 1.0f / d;
    dis[i]  = rsqrtf(d);
  }
}

// ---- fp32 -> bf16 weight pre-conversion ------------------------------------
__global__ void k_cvt_bf16(const float* __restrict__ src, __bf16* __restrict__ dst, int n) {
  int i = blockIdx.x * blockDim.x + threadIdx.x;
  if (i < n) dst[i] = to_bf(src[i]);
}

// ---- WMMA GEMM: C[M,128] = A[M,K] * W[128,K]^T (+bias), optional 2nd copy --
// Block = 256 threads = 8 waves; block handles one 16-row M tile, wave wv
// handles N tile wv (H = 128 = 8 * 16).
//
// A operand (16-bit A 16x32, ISA 7.12.2): lane m = lane&15, group g = lane>>4,
//   element 4p+t of the v16bf  <->  A[m][ cc*32 + g*8 + (p&1)*4 + (p>>1)*16 + t ]
// We pre-gather that exact layout into LDS: slot (cc, lane) holds the lane's
// 16 bf16 (32 B) for K-chunk cc, so compute waves read it with one 32 B load.
//
// B operand (32x16): lane n = lane&15, group g, element 4p+t <->
//   B[cc*32 + g*8 + (p&1)*4 + (p>>1)*16 + t][n] = W[n][...]  (direct bf16 global loads)
//
// C layout: acc[r] -> C[mtile*16 + r + 8*g][wv*16 + (lane&15)]
template<int K, bool HAS_BIAS, bool DUP>
__global__ void __launch_bounds__(256)
k_wmma_gemm(const float* __restrict__ A,
            const __bf16* __restrict__ Wbf,
            const float* __restrict__ bias,
            float* __restrict__ C0,
            float* __restrict__ C1) {
  constexpr int CHUNKS = K / 32;
  constexpr int SLOTS  = CHUNKS * 32;
  __shared__ __align__(32) __bf16 smA[SLOTS * 16];

  const int tid   = threadIdx.x;
  const int lane  = tid & 31;
  const int wv    = tid >> 5;                 // 0..7 : N tile
  const int g     = lane >> 4;
  const int l16   = lane & 15;
  const int mbase = blockIdx.x * 16;

  // ---- stage A tile into LDS in WMMA operand order, converting to bf16 ----
  if (tid < SLOTS) {
    const int cc = tid >> 5;                  // K chunk
    const int sl = tid & 31;                  // lane slot
    const int sg = sl >> 4;
    const float* __restrict__ Arow = A + (size_t)(mbase + (sl & 15)) * K;
    v16bf tmp;
#pragma unroll
    for (int p = 0; p < 4; ++p) {
      const int ka = cc * 32 + sg * 8 + (p & 1) * 4 + (p >> 1) * 16;
      const float4 af = *(const float4*)(Arow + ka);
      tmp[4 * p + 0] = to_bf(af.x);
      tmp[4 * p + 1] = to_bf(af.y);
      tmp[4 * p + 2] = to_bf(af.z);
      tmp[4 * p + 3] = to_bf(af.w);
    }
    *(v16bf*)(&smA[(size_t)tid * 16]) = tmp;  // 2x ds_store_b128
  }
  __syncthreads();

  // ---- WMMA main loop (full unroll: CHUNKS wmma ops, EXEC all-ones) -------
  const int nrow = wv * 16 + l16;
  const __bf16* __restrict__ Wrow = Wbf + (size_t)nrow * K;

  v8f acc = {};
#pragma unroll
  for (int cc = 0; cc < CHUNKS; ++cc) {
    const v16bf a = *(const v16bf*)(&smA[(size_t)(cc * 32 + lane) * 16]);
    v16bf b;
#pragma unroll
    for (int p = 0; p < 4; ++p) {
      const int kb = cc * 32 + g * 8 + (p & 1) * 4 + (p >> 1) * 16;
      const bf4 wf = *(const bf4*)(Wrow + kb);
      b[4 * p + 0] = wf[0];
      b[4 * p + 1] = wf[1];
      b[4 * p + 2] = wf[2];
      b[4 * p + 3] = wf[3];
    }
    acc = __builtin_amdgcn_wmma_f32_16x16x32_bf16(
        /*neg_a=*/false, a, /*neg_b=*/false, b,
        /*c_mod=*/(short)0, acc, /*reuse_a=*/false, /*reuse_b=*/false);
  }

  // ---- branch-free epilogue ----
  float badd = 0.0f;
  if (HAS_BIAS) badd = bias[nrow];
#pragma unroll
  for (int r = 0; r < 8; ++r) {
    const int mr = mbase + g * 8 + r;
    const float v = acc[r] + badd;
    C0[(size_t)mr * Hc + nrow] = v;
    if (DUP) C1[(size_t)mr * Hc + nrow] = v;
  }
}

// ---- residual + LayerNorm + ReLU (one wave per row, H=128 = 4 per lane) ----
__global__ void k_pre(float* __restrict__ h, const float* __restrict__ ori,
                      const float* __restrict__ gamma, const float* __restrict__ beta,
                      int n) {
  const int wid  = (blockIdx.x * blockDim.x + threadIdx.x) >> 5;
  const int lane = threadIdx.x & 31;
  if (wid >= n) return;
  float* __restrict__ hr = h + (size_t)wid * Hc;
  const float* __restrict__ orow = ori + (size_t)wid * Hc;

  float x[4], s = 0.0f;
#pragma unroll
  for (int t = 0; t < 4; ++t) {
    const int j = lane + 32 * t;
    x[t] = hr[j] + orow[j];
    s += x[t];
  }
#pragma unroll
  for (int off = 16; off > 0; off >>= 1) s += __shfl_xor(s, off, 32);
  const float mu = s * (1.0f / Hc);

  float vs = 0.0f;
#pragma unroll
  for (int t = 0; t < 4; ++t) { const float d = x[t] - mu; vs += d * d; }
#pragma unroll
  for (int off = 16; off > 0; off >>= 1) vs += __shfl_xor(vs, off, 32);
  const float inv = rsqrtf(vs * (1.0f / Hc) + LN_EPS);

#pragma unroll
  for (int t = 0; t < 4; ++t) {
    const int j = lane + 32 * t;
    const float y = (x[t] - mu) * inv * gamma[j] + beta[j];
    hr[j] = fmaxf(y, 0.0f);
  }
}

// ---- h_msg init: conv_b + self-loop contribution deg_inv[n]*xw[n] ----------
__global__ void k_init_msg(const float* __restrict__ xw, const float* __restrict__ dinv,
                           const float* __restrict__ conv_b, float* __restrict__ hmsg,
                           int total) {
  const int idx = blockIdx.x * blockDim.x + threadIdx.x;
  if (idx < total) {
    const int j    = idx & (Hc - 1);
    const int node = idx >> 7;
    hmsg[idx] = conv_b[j] + dinv[node] * xw[idx];
  }
}

// ---- edge scatter: one wave per edge, 4 floats per lane, L2-resident atomics
__global__ void k_scatter(const int* __restrict__ row, const int* __restrict__ col,
                          const float* __restrict__ dis, const float* __restrict__ xw,
                          float* __restrict__ hmsg, int e) {
  const int wid  = (blockIdx.x * blockDim.x + threadIdx.x) >> 5;
  const int lane = threadIdx.x & 31;
  if (wid >= e) return;
  const int r = row[wid];
  const int c = col[wid];
  const float w = dis[r] * dis[c];
  const float4 v = ((const float4*)(xw + (size_t)c * Hc))[lane];
  float* __restrict__ dst = hmsg + (size_t)r * Hc + lane * 4;
  atomicAdd(dst + 0, w * v.x);
  atomicAdd(dst + 1, w * v.y);
  atomicAdd(dst + 2, w * v.z);
  atomicAdd(dst + 3, w * v.w);
}

// ---- combine: dst = h_msg + relu(h + root) * deg_inv (dst may alias h_msg) -
__global__ void k_combine(const float* __restrict__ hmsg, const float* __restrict__ h,
                          const float* __restrict__ root, const float* __restrict__ dinv,
                          float* __restrict__ dst, int total) {
  const int idx = blockIdx.x * blockDim.x + threadIdx.x;
  if (idx < total) {
    const int j    = idx & (Hc - 1);
    const int node = idx >> 7;
    dst[idx] = hmsg[idx] + fmaxf(h[idx] + root[j], 0.0f) * dinv[node];
  }
}

// ---------------------------------------------------------------------------
extern "C" void kernel_launch(void* const* d_in, const int* in_sizes, int n_in,
                              void* d_out, int out_size, void* d_ws, size_t ws_size,
                              hipStream_t stream) {
  const float* in_feat = (const float*)d_in[0];
  const int*   row     = (const int*)  d_in[1];
  const int*   col     = (const int*)  d_in[2];
  const float* lin_w   = (const float*)d_in[3];
  const float* lin_b   = (const float*)d_in[4];
  const float* conv_w  = (const float*)d_in[5];
  const float* conv_b  = (const float*)d_in[6];
  const float* root    = (const float*)d_in[7];
  const float* ln_g    = (const float*)d_in[8];
  const float* ln_b    = (const float*)d_in[9];
  float*       out     = (float*)d_out;

  const int N = in_sizes[0] / INc;     // 50000 (multiple of 16)
  const int E = in_sizes[1];           // 600000
  const int total = N * Hc;

  float*  ws     = (float*)d_ws;
  float*  ori    = ws;                      // [N,H] f32
  float*  h      = ori  + (size_t)N * Hc;   // [N,H]
  float*  xw     = h    + (size_t)N * Hc;   // [N,H]
  float*  deg    = xw   + (size_t)N * Hc;   // [N]
  float*  dis    = deg  + N;                // [N]
  float*  dinv   = dis  + N;                // [N]
  __bf16* linwbf = (__bf16*)(dinv + N);     // [H,IN] bf16 (64 KB)
  __bf16* convwbf= linwbf + (size_t)Hc * INc; // [H,H] bf16 (32 KB)

  // weight pre-conversion to bf16 (once per launch)
  k_cvt_bf16<<<(Hc * INc + 255) / 256, 256, 0, stream>>>(lin_w,  linwbf,  Hc * INc);
  k_cvt_bf16<<<(Hc * Hc  + 255) / 256, 256, 0, stream>>>(conv_w, convwbf, Hc * Hc);

  // degrees (d2 == deg thanks to the +1 self-loop identity)
  k_init_deg  <<<(N + 255) / 256, 256, 0, stream>>>(deg, N);
  k_count_deg <<<(E + 255) / 256, 256, 0, stream>>>(row, deg, E);
  k_finish_deg<<<(N + 255) / 256, 256, 0, stream>>>(deg, dis, dinv, N);

  const int mtiles = N / 16;

  // h = ori = in_feat @ lin_w.T + lin_b   (WMMA bf16, f32 acc)
  k_wmma_gemm<INc, true, true><<<mtiles, 256, 0, stream>>>(in_feat, linwbf, lin_b, h, ori);

  for (int step = 0; step < 2; ++step) {
    if (step)
      k_pre<<<(N * 32 + 255) / 256, 256, 0, stream>>>(h, ori, ln_g + Hc, ln_b + Hc, N);

    // xw = h @ conv_w.T
    k_wmma_gemm<Hc, false, false><<<mtiles, 256, 0, stream>>>(h, convwbf, nullptr, xw, nullptr);

    // h_msg accumulates in d_out (fully rewritten each call -> deterministic)
    k_init_msg<<<(total + 255) / 256, 256, 0, stream>>>(xw, dinv, conv_b, out, total);
    k_scatter <<<(E * 32 + 255) / 256, 256, 0, stream>>>(row, col, dis, xw, out, E);

    float* dst = (step == 1) ? out : h;   // last step finalizes d_out in place
    k_combine<<<(total + 255) / 256, 256, 0, stream>>>(out, h, root, dinv, dst, total);
  }
}
